// TransformerEncoderLayer_36558761623882
// MI455X (gfx1250) — matrix operand order repaired and checked by reading, compiled, not run on hardware
//
#include <hip/hip_runtime.h>
#include <math.h>

typedef __bf16 bf16_t;
typedef __attribute__((ext_vector_type(16))) __bf16 v16bf;
typedef __attribute__((ext_vector_type(8)))  __bf16 v8bf;
typedef __attribute__((ext_vector_type(8)))  float  v8f;

constexpr int Bn   = 4;
constexpr int Ln   = 20000;
constexpr int Dn   = 256;
constexpr int Hn   = 8;
constexpr int DFFn = 1024;
constexpr int Kn   = 1000;
constexpr int Nn   = Bn * Ln;    // 80000
constexpr int NSEL = Bn * Kn;    // 4000

__device__ __forceinline__ v8f zero8() {
  v8f z = {0.f,0.f,0.f,0.f,0.f,0.f,0.f,0.f};
  return z;
}

// assemble a 16-element bf16 fragment from two 16-byte chunks
__device__ __forceinline__ v16bf frag2(const bf16_t* p0, const bf16_t* p1) {
  v8bf lo = *(const v8bf*)p0;
  v8bf hi = *(const v8bf*)p1;
  v16bf r;
#pragma unroll
  for (int j = 0; j < 8; ++j) { r[j] = lo[j]; r[j + 8] = hi[j]; }
  return r;
}

__device__ __forceinline__ v8f wmma_bf(v16bf a, v16bf b, v8f c) {
  return __builtin_amdgcn_wmma_f32_16x16x32_bf16(false, a, false, b, (short)0, c, false, false);
}

// ---------------------------------------------------------------- utilities
__global__ __launch_bounds__(256) void copy_f32x4(const float* __restrict__ s,
                                                  float* __restrict__ d, long n4) {
  long i = (long)blockIdx.x * blockDim.x + threadIdx.x;
  if (i < n4) ((float4*)d)[i] = ((const float4*)s)[i];
}

__global__ __launch_bounds__(256) void cvt_bf16(const float* __restrict__ s,
                                                bf16_t* __restrict__ d, int n) {
  int i = blockIdx.x * blockDim.x + threadIdx.x;
  if (i < n) d[i] = (bf16_t)s[i];
}

// -------------------------------------------------------- per-batch top-K
__global__ __launch_bounds__(1024) void topk_kernel(const float* __restrict__ sal,
                                                    const float* __restrict__ prob,
                                                    int* __restrict__ fidx) {
  __shared__ unsigned hist[2048];
  __shared__ unsigned sT1, sC1, sT2, sC2, gcnt, tcnt;
  int b = blockIdx.x, t = threadIdx.x;

  auto key_of = [&](int i) -> unsigned {
    int gi = b * Ln + i;
    float s = prob[gi] * (1.f / (1.f + __expf(-sal[gi])));
    unsigned u = __float_as_uint(s);
    u ^= (u >> 31) ? 0xFFFFFFFFu : 0x80000000u;
    return u;
  };

  for (int i = t; i < 2048; i += 1024) hist[i] = 0;
  __syncthreads();
  for (int i = t; i < Ln; i += 1024) atomicAdd(&hist[key_of(i) >> 21], 1u);
  __syncthreads();
  if (t == 0) {
    unsigned acc = 0; int bin = 2047;
    for (; bin > 0; --bin) { unsigned c = hist[bin]; if (acc + c >= (unsigned)Kn) break; acc += c; }
    sT1 = (unsigned)bin; sC1 = acc;
  }
  __syncthreads();
  unsigned T1 = sT1, C1 = sC1;
  for (int i = t; i < 2048; i += 1024) hist[i] = 0;
  __syncthreads();
  for (int i = t; i < Ln; i += 1024) {
    unsigned k = key_of(i);
    if ((k >> 21) == T1) atomicAdd(&hist[(k >> 10) & 2047u], 1u);
  }
  __syncthreads();
  if (t == 0) {
    unsigned acc = C1; int bin = 2047;
    for (; bin > 0; --bin) { unsigned c = hist[bin]; if (acc + c >= (unsigned)Kn) break; acc += c; }
    sT2 = (unsigned)bin; sC2 = acc; gcnt = 0; tcnt = 0;
  }
  __syncthreads();
  unsigned thresh = (T1 << 11) | sT2, cntHi = sC2;
  for (int i = t; i < Ln; i += 1024) {
    unsigned k = key_of(i) >> 10;
    if (k > thresh) {
      unsigned p = atomicAdd(&gcnt, 1u);
      if (p < (unsigned)Kn) fidx[b * Kn + p] = b * Ln + i;
    } else if (k == thresh) {
      unsigned p = atomicAdd(&tcnt, 1u);
      if (cntHi + p < (unsigned)Kn) fidx[b * Kn + cntHi + p] = b * Ln + i;
    }
  }
}

// --------------------------------------------- gather selected rows + LN1
__global__ __launch_bounds__(256) void gather_ln1(const float* __restrict__ q,
                                                  const float* __restrict__ pos,
                                                  const int* __restrict__ fidx,
                                                  const float* __restrict__ g,
                                                  const float* __restrict__ bt,
                                                  bf16_t* __restrict__ xn,
                                                  bf16_t* __restrict__ qkin) {
  int wave = threadIdx.x >> 5, lane = threadIdx.x & 31;
  int row = blockIdx.x * 8 + wave;
  if (row >= NSEL) return;
  int src = fidx[row];
  const float* xr = q + (long)src * Dn;
  const float* pr = pos + (long)src * Dn;
  float v[8], s = 0.f, s2 = 0.f;
#pragma unroll
  for (int j = 0; j < 8; ++j) { float x = xr[lane + 32 * j]; v[j] = x; s += x; s2 += x * x; }
  for (int o = 16; o > 0; o >>= 1) { s += __shfl_xor(s, o, 32); s2 += __shfl_xor(s2, o, 32); }
  float mean = s / Dn, var = s2 / Dn - mean * mean;
  float rr = rsqrtf(var + 1e-5f);
#pragma unroll
  for (int j = 0; j < 8; ++j) {
    int c = lane + 32 * j;
    float xh = (v[j] - mean) * rr * g[c] + bt[c];
    xn[(long)row * Dn + c]   = (bf16_t)xh;
    qkin[(long)row * Dn + c] = (bf16_t)(xh + pr[c]);
  }
}

// ------------------------------ C[M x 256] = A[M x 256] * W[256 x 256], bf16
__global__ __launch_bounds__(256) void gemm_nt(const bf16_t* __restrict__ A,
                                               const bf16_t* __restrict__ Wb,
                                               bf16_t* __restrict__ C, int M) {
  __shared__ __align__(16) bf16_t As[128][264];
  int wave = threadIdx.x >> 5, lane = threadIdx.x & 31;
  int mlane = lane & 15, kh = lane >> 4;
  int rowbase = blockIdx.x * 128;
  for (int i = 0; i < 16; ++i) {
    int r = i * 8 + wave;
    int grow = rowbase + r; if (grow >= M) grow = M - 1;
    const bf16_t* src = A + (long)grow * Dn;
#pragma unroll
    for (int j = 0; j < 8; ++j) As[r][lane + 32 * j] = src[lane + 32 * j];
  }
  __syncthreads();
  int m0 = wave * 16;
  v16bf afr[8];
#pragma unroll
  for (int kb = 0; kb < 8; ++kb) {
    const bf16_t* p = &As[m0 + mlane][kb * 32 + kh * 8];
    afr[kb] = frag2(p, p + 16);
  }
  v8f acc[16];
#pragma unroll
  for (int n = 0; n < 16; ++n) acc[n] = zero8();
#pragma unroll
  for (int n = 0; n < 16; ++n) {
#pragma unroll
    for (int kb = 0; kb < 8; ++kb) {
      v16bf bfr;
#pragma unroll
      for (int s = 0; s < 16; ++s) bfr[s] = Wb[(long)(kb * 32 + kh * 8 + (s & 7) + (s >> 3) * 16) * Dn + n * 16 + mlane];
      acc[n] = wmma_bf(afr[kb], bfr, acc[n]);
    }
  }
#pragma unroll
  for (int n = 0; n < 16; ++n)
#pragma unroll
    for (int r = 0; r < 8; ++r) {
      int grow = rowbase + m0 + r + 8 * kh;
      if (grow < M) C[(long)grow * Dn + n * 16 + mlane] = (bf16_t)acc[n][r];
    }
}

// --------------------------- O @ Wo + residual, scattered through flat_idx
__global__ __launch_bounds__(256) void gemm_o_scatter(const bf16_t* __restrict__ A,
                                                      const bf16_t* __restrict__ Wb,
                                                      const int* __restrict__ fidx,
                                                      float* __restrict__ out, int M) {
  __shared__ __align__(16) bf16_t As[128][264];
  int wave = threadIdx.x >> 5, lane = threadIdx.x & 31;
  int mlane = lane & 15, kh = lane >> 4;
  int rowbase = blockIdx.x * 128;
  for (int i = 0; i < 16; ++i) {
    int r = i * 8 + wave;
    int grow = rowbase + r; if (grow >= M) grow = M - 1;
    const bf16_t* src = A + (long)grow * Dn;
#pragma unroll
    for (int j = 0; j < 8; ++j) As[r][lane + 32 * j] = src[lane + 32 * j];
  }
  __syncthreads();
  int m0 = wave * 16;
  v16bf afr[8];
#pragma unroll
  for (int kb = 0; kb < 8; ++kb) {
    const bf16_t* p = &As[m0 + mlane][kb * 32 + kh * 8];
    afr[kb] = frag2(p, p + 16);
  }
  v8f acc[16];
#pragma unroll
  for (int n = 0; n < 16; ++n) acc[n] = zero8();
#pragma unroll
  for (int n = 0; n < 16; ++n) {
#pragma unroll
    for (int kb = 0; kb < 8; ++kb) {
      v16bf bfr;
#pragma unroll
      for (int s = 0; s < 16; ++s) bfr[s] = Wb[(long)(kb * 32 + kh * 8 + (s & 7) + (s >> 3) * 16) * Dn + n * 16 + mlane];
      acc[n] = wmma_bf(afr[kb], bfr, acc[n]);
    }
  }
#pragma unroll
  for (int n = 0; n < 16; ++n) {
    int col = n * 16 + mlane;
#pragma unroll
    for (int r = 0; r < 8; ++r) {
      int grow = rowbase + m0 + r + 8 * kh;
      if (grow < M) {
        int drow = fidx[grow];
        float* o = out + (long)drow * Dn + col;
        *o = *o + acc[n][r];
      }
    }
  }
}

// -------------------------------- flash attention over selected tokens
__global__ __launch_bounds__(256) void attn_kernel(const bf16_t* __restrict__ Qb,
                                                   const bf16_t* __restrict__ Kb,
                                                   const bf16_t* __restrict__ Vb,
                                                   bf16_t* __restrict__ Ob) {
  constexpr float SCALE = 0.17677669529663687f;  // 1/sqrt(32)
  __shared__ __align__(16) bf16_t Kt[32][40];        // [key][dh]
  __shared__ __align__(16) bf16_t Vs[32][40];        // [dh][key]
  __shared__ __align__(16) bf16_t Ps[8][16][40];     // per-wave P staging
  int bh = blockIdx.x;
  int b = bh >> 3, h = bh & 7;
  int wave = threadIdx.x >> 5, lane = threadIdx.x & 31;
  int mlane = lane & 15, kh = lane >> 4;
  int q0 = blockIdx.y * 128 + wave * 16;

  int qr = q0 + mlane; if (qr >= Kn) qr = Kn - 1;
  const bf16_t* qp = Qb + (long)(b * Kn + qr) * Dn + h * 32 + kh * 8;
  v16bf qfr = frag2(qp, qp + 16);

  v8f o0 = zero8(), o1 = zero8();
  float mst[8], lst[8];
#pragma unroll
  for (int r = 0; r < 8; ++r) { mst[r] = -1e30f; lst[r] = 0.f; }

  for (int c0 = 0; c0 < Kn; c0 += 32) {
    for (int e = threadIdx.x; e < 1024; e += 256) {
      int kk = e >> 5, dd = e & 31;
      int krow = c0 + kk;
      bf16_t kv = (bf16_t)0.f, vv = (bf16_t)0.f;
      if (krow < Kn) {
        kv = Kb[(long)(b * Kn + krow) * Dn + h * 32 + dd];
        vv = Vb[(long)(b * Kn + krow) * Dn + h * 32 + dd];
      }
      Kt[kk][dd] = kv;
      Vs[dd][kk] = vv;
    }
    __syncthreads();

    float sv0[8], sv1[8];
    {
      v16bf bk0 = frag2(&Kt[mlane][kh * 8],  &Kt[mlane][16 + kh * 8]);
      v16bf bk1 = frag2(&Kt[16 + mlane][kh * 8], &Kt[16 + mlane][16 + kh * 8]);
      v8f s0 = wmma_bf(qfr, bk0, zero8());
      v8f s1 = wmma_bf(qfr, bk1, zero8());
      float mask0 = (c0 + mlane      < Kn) ? 0.f : -1e30f;
      float mask1 = (c0 + 16 + mlane < Kn) ? 0.f : -1e30f;
#pragma unroll
      for (int r = 0; r < 8; ++r) {
        sv0[r] = s0[r] * SCALE + mask0;
        sv1[r] = s1[r] * SCALE + mask1;
      }
    }
    // online softmax, per row (rows replicated across 16 lanes of each half)
#pragma unroll
    for (int r = 0; r < 8; ++r) {
      float cm = fmaxf(sv0[r], sv1[r]);
      for (int o = 8; o > 0; o >>= 1) cm = fmaxf(cm, __shfl_xor(cm, o, 32));
      float newm = fmaxf(mst[r], cm);
      float alpha = __expf(mst[r] - newm);
      float p0 = __expf(sv0[r] - newm);
      float p1 = __expf(sv1[r] - newm);
      float ps = p0 + p1;
      for (int o = 8; o > 0; o >>= 1) ps += __shfl_xor(ps, o, 32);
      lst[r] = lst[r] * alpha + ps;
      mst[r] = newm;
      o0[r] *= alpha; o1[r] *= alpha;
      Ps[wave][r + 8 * kh][mlane]      = (bf16_t)p0;
      Ps[wave][r + 8 * kh][16 + mlane] = (bf16_t)p1;
    }
    asm volatile("s_wait_dscnt 0" ::: "memory");

    const bf16_t* pp = &Ps[wave][mlane][kh * 8];
    v16bf pfr = frag2(pp, pp + 16);
    v16bf vf0 = frag2(&Vs[mlane][kh * 8],  &Vs[mlane][16 + kh * 8]);
    v16bf vf1 = frag2(&Vs[16 + mlane][kh * 8], &Vs[16 + mlane][16 + kh * 8]);
    o0 = wmma_bf(pfr, vf0, o0);
    o1 = wmma_bf(pfr, vf1, o1);
    __syncthreads();
  }
#pragma unroll
  for (int r = 0; r < 8; ++r) {
    int qq = q0 + r + 8 * kh;
    if (qq < Kn) {
      float inv = 1.f / lst[r];
      long row = (long)(b * Kn + qq) * Dn + h * 32;
      Ob[row + mlane]      = (bf16_t)(o0[r] * inv);
      Ob[row + 16 + mlane] = (bf16_t)(o1[r] * inv);
    }
  }
}

// -------------------------------- fused LN2 + FFN over all tokens, in place
__global__ __launch_bounds__(256) void ffn_kernel(float* __restrict__ X,
                                                  const bf16_t* __restrict__ W1b,
                                                  const bf16_t* __restrict__ W2b,
                                                  const float* __restrict__ b1,
                                                  const float* __restrict__ b2,
                                                  const float* __restrict__ g,
                                                  const float* __restrict__ bt, int M) {
  __shared__ __align__(16) bf16_t Hs[128][264];
  __shared__ __align__(16) bf16_t Gs[8][16][80];
  int wave = threadIdx.x >> 5, lane = threadIdx.x & 31;
  int mlane = lane & 15, kh = lane >> 4;
  int rowbase = blockIdx.x * 128;

  // LN2 into bf16 LDS tile
  for (int i = 0; i < 16; ++i) {
    int r = i * 8 + wave;
    long grow = rowbase + r; if (grow >= M) grow = M - 1;
    const float* xr = X + grow * Dn;
    float v[8], s = 0.f, s2 = 0.f;
#pragma unroll
    for (int j = 0; j < 8; ++j) { float x = xr[lane + 32 * j]; v[j] = x; s += x; s2 += x * x; }
    for (int o = 16; o > 0; o >>= 1) { s += __shfl_xor(s, o, 32); s2 += __shfl_xor(s2, o, 32); }
    float mean = s / Dn, var = s2 / Dn - mean * mean;
    float rr = rsqrtf(var + 1e-5f);
#pragma unroll
    for (int j = 0; j < 8; ++j) {
      int c = lane + 32 * j;
      Hs[r][c] = (bf16_t)((v[j] - mean) * rr * g[c] + bt[c]);
    }
  }
  __syncthreads();

  int m0 = wave * 16;
  v16bf afr[8];
#pragma unroll
  for (int kb = 0; kb < 8; ++kb) {
    const bf16_t* p = &Hs[m0 + mlane][kb * 32 + kh * 8];
    afr[kb] = frag2(p, p + 16);
  }
  v8f acc[16];
#pragma unroll
  for (int n = 0; n < 16; ++n) acc[n] = zero8();

  for (int ch = 0; ch < 16; ++ch) {          // 16 chunks of 64 ff columns
    v8f tt[4];
#pragma unroll
    for (int nt = 0; nt < 4; ++nt) {
      v8f a = zero8();
#pragma unroll
      for (int kb = 0; kb < 8; ++kb) {
        v16bf bfr;
#pragma unroll
        for (int s = 0; s < 16; ++s) bfr[s] = W1b[(long)(kb * 32 + kh * 8 + (s & 7) + (s >> 3) * 16) * DFFn + ch * 64 + nt * 16 + mlane];
        a = wmma_bf(afr[kb], bfr, a);
      }
      tt[nt] = a;
    }
    // bias + exact GELU, restage to A-layout LDS
#pragma unroll
    for (int nt = 0; nt < 4; ++nt) {
      float bb = b1[ch * 64 + nt * 16 + mlane];
#pragma unroll
      for (int r = 0; r < 8; ++r) {
        float x = tt[nt][r] + bb;
        float ge = 0.5f * x * (1.f + erff(x * 0.70710678118654752f));
        Gs[wave][r + 8 * kh][nt * 16 + mlane] = (bf16_t)ge;
      }
    }
    asm volatile("s_wait_dscnt 0" ::: "memory");
    const bf16_t* g0p = &Gs[wave][mlane][kh * 8];
    v16bf g0 = frag2(g0p, g0p + 16);
    const bf16_t* g1p = &Gs[wave][mlane][32 + kh * 8];
    v16bf g1 = frag2(g1p, g1p + 16);
#pragma unroll
    for (int n = 0; n < 16; ++n) {
      v16bf w0, w1;
#pragma unroll
      for (int s = 0; s < 16; ++s) {
        int kk = kh * 8 + (s & 7) + (s >> 3) * 16;
        w0[s] = W2b[(long)(ch * 64 + kk) * Dn + n * 16 + mlane];
        w1[s] = W2b[(long)(ch * 64 + 32 + kk) * Dn + n * 16 + mlane];
      }
      acc[n] = wmma_bf(g0, w0, acc[n]);
      acc[n] = wmma_bf(g1, w1, acc[n]);
    }
  }
  // epilogue: out = x + acc + b2
#pragma unroll
  for (int n = 0; n < 16; ++n) {
    int col = n * 16 + mlane;
    float b2v = b2[col];
#pragma unroll
    for (int r = 0; r < 8; ++r) {
      long grow = rowbase + m0 + r + 8 * kh;
      if (grow < M) {
        float* o = X + grow * Dn + col;
        *o = *o + acc[n][r] + b2v;
      }
    }
  }
}

// ---------------------------------------------------------------- launcher
extern "C" void kernel_launch(void* const* d_in, const int* in_sizes, int n_in,
                              void* d_out, int out_size, void* d_ws, size_t ws_size,
                              hipStream_t stream) {
  (void)in_sizes; (void)n_in; (void)out_size; (void)ws_size;
  const float* queries = (const float*)d_in[0];
  const float* qpos    = (const float*)d_in[1];
  const float* sal     = (const float*)d_in[7];
  const float* prob    = (const float*)d_in[8];
  const float* ln1g    = (const float*)d_in[9];
  const float* ln1b    = (const float*)d_in[10];
  const float* Wq      = (const float*)d_in[11];
  const float* Wk      = (const float*)d_in[12];
  const float* Wv      = (const float*)d_in[13];
  const float* Wo      = (const float*)d_in[14];
  const float* ln2g    = (const float*)d_in[15];
  const float* ln2b    = (const float*)d_in[16];
  const float* W1      = (const float*)d_in[17];
  const float* b1      = (const float*)d_in[18];
  const float* W2      = (const float*)d_in[19];
  const float* b2      = (const float*)d_in[20];
  float* out = (float*)d_out;

  char* ws = (char*)d_ws;
  size_t off = 0;
  auto take = [&](size_t bytes) -> char* {
    char* p = ws + off;
    off += (bytes + 255) & ~(size_t)255;
    return p;
  };
  int*    fidx = (int*)   take(NSEL * sizeof(int));
  bf16_t* xnb  = (bf16_t*)take((size_t)NSEL * Dn * 2);
  bf16_t* qkb  = (bf16_t*)take((size_t)NSEL * Dn * 2);
  bf16_t* Qbf  = (bf16_t*)take((size_t)NSEL * Dn * 2);
  bf16_t* Kbf  = (bf16_t*)take((size_t)NSEL * Dn * 2);
  bf16_t* Vbf  = (bf16_t*)take((size_t)NSEL * Dn * 2);
  bf16_t* Obf  = (bf16_t*)take((size_t)NSEL * Dn * 2);
  bf16_t* Wqb  = (bf16_t*)take((size_t)Dn * Dn * 2);
  bf16_t* Wkb  = (bf16_t*)take((size_t)Dn * Dn * 2);
  bf16_t* Wvb  = (bf16_t*)take((size_t)Dn * Dn * 2);
  bf16_t* Wob  = (bf16_t*)take((size_t)Dn * Dn * 2);
  bf16_t* W1b  = (bf16_t*)take((size_t)Dn * DFFn * 2);
  bf16_t* W2b  = (bf16_t*)take((size_t)DFFn * Dn * 2);

  long n4 = (long)Nn * Dn / 4;
  copy_f32x4<<<(int)((n4 + 255) / 256), 256, 0, stream>>>(queries, out, n4);

  cvt_bf16<<<(Dn * Dn + 255) / 256, 256, 0, stream>>>(Wq, Wqb, Dn * Dn);
  cvt_bf16<<<(Dn * Dn + 255) / 256, 256, 0, stream>>>(Wk, Wkb, Dn * Dn);
  cvt_bf16<<<(Dn * Dn + 255) / 256, 256, 0, stream>>>(Wv, Wvb, Dn * Dn);
  cvt_bf16<<<(Dn * Dn + 255) / 256, 256, 0, stream>>>(Wo, Wob, Dn * Dn);
  cvt_bf16<<<(Dn * DFFn + 255) / 256, 256, 0, stream>>>(W1, W1b, Dn * DFFn);
  cvt_bf16<<<(DFFn * Dn + 255) / 256, 256, 0, stream>>>(W2, W2b, DFFn * Dn);

  topk_kernel<<<Bn, 1024, 0, stream>>>(sal, prob, fidx);
  gather_ln1<<<NSEL / 8, 256, 0, stream>>>(queries, qpos, fidx, ln1g, ln1b, xnb, qkb);

  int gblk = (NSEL + 127) / 128;
  gemm_nt<<<gblk, 256, 0, stream>>>(qkb, Wqb, Qbf, NSEL);
  gemm_nt<<<gblk, 256, 0, stream>>>(qkb, Wkb, Kbf, NSEL);
  gemm_nt<<<gblk, 256, 0, stream>>>(xnb, Wvb, Vbf, NSEL);

  attn_kernel<<<dim3(Bn * Hn, 8), 256, 0, stream>>>(Qbf, Kbf, Vbf, Obf);
  gemm_o_scatter<<<gblk, 256, 0, stream>>>(Obf, Wob, fidx, out, NSEL);

  ffn_kernel<<<Nn / 128, 256, 0, stream>>>(out, W1b, W2b, b1, b2, ln2g, ln2b, Nn);
}
